// MultiPostProcessingMLP_24223615549692
// MI455X (gfx1250) — matrix operand order repaired and checked by reading, not compiled, not run on hardware
//
#include <hip/hip_runtime.h>
#include <hip/hip_bf16.h>

typedef __attribute__((ext_vector_type(16))) _Float16 v16h;
typedef __attribute__((ext_vector_type(8)))  _Float16 v8h;
typedef __attribute__((ext_vector_type(8)))  float    v8f;
typedef __attribute__((ext_vector_type(4)))  int      v4i;

#define KEXP  8
#define WIDTH 64
#define DFEAT 32
#define DIN   74
#define DINP  96     // 74 padded to multiple of 32 (WMMA K step)
#define TROWS 128    // rows per tile / block

#if __has_builtin(__builtin_amdgcn_global_load_async_to_lds_b128)
#define ATHENA_ASYNC_LDS 1
#else
#define ATHENA_ASYNC_LDS 0
#endif

union F16F { v16h v; v8h h8[2]; };

// A-fragment (16x32 f16): lane (l&15)=row M; halves [kb + (l>>4)*8 .. +7] and [+16 .. +23]
static __device__ inline v16h load_frag_A(const _Float16* rowp, int kb, int rhi) {
  F16F f;
  f.h8[0] = *(const v8h*)(rowp + kb + rhi * 8);
  f.h8[1] = *(const v8h*)(rowp + kb + rhi * 8 + 16);
  return f.v;
}
// B-fragment (32x16 f16) from W stored [dout][din]: lane (l&15)=col N;
// halves [kb + (l>>4)*8 .. +7] and [+16 .. +23]
static __device__ inline v16h load_frag_B(const _Float16* wbase, int stride, int n, int kb, int rhi) {
  const _Float16* p = wbase + n * stride + kb + rhi * 8;
  F16F f;
  f.h8[0] = *(const v8h*)(p);
  f.h8[1] = *(const v8h*)(p + 16);
  return f.v;
}

#if ATHENA_ASYNC_LDS
// 16-byte async global->LDS copy (tracked by ASYNCcnt, no VGPR round-trip)
static __device__ inline void async_copy16(const void* g, void* l) {
  __builtin_amdgcn_global_load_async_to_lds_b128(
      (__attribute__((address_space(1))) v4i*)(g),
      (__attribute__((address_space(3))) v4i*)(l), 0, 0);
}
static __device__ inline void wait_async0() {
#if __has_builtin(__builtin_amdgcn_s_wait_asynccnt)
  __builtin_amdgcn_s_wait_asynccnt(0);
#else
  asm volatile("s_wait_asynccnt 0x0" ::: "memory");
#endif
}
#endif

// ---------------- preprocessing kernels ----------------

__global__ void zero_kernel(int* counters) {
  if (threadIdx.x < 16) counters[threadIdx.x] = 0;   // counts[8] + cursors[8]
}

__global__ void count_kernel(const int* __restrict__ idxs, int* __restrict__ counts, int n) {
  int i = blockIdx.x * 256 + threadIdx.x;
  if (i < n) atomicAdd(&counts[idxs[i]], 1);
}

__global__ void offsets_kernel(const int* __restrict__ counts, int* __restrict__ offsets,
                               int* __restrict__ tileExpert, int maxTiles) {
  __shared__ int soff[KEXP + 1];
  if (threadIdx.x == 0) {
    int acc = 0;
    for (int k = 0; k < KEXP; ++k) {
      soff[k] = acc; offsets[k] = acc;
      acc += ((counts[k] + TROWS - 1) / TROWS) * TROWS;
    }
    soff[KEXP] = acc; offsets[KEXP] = acc;
  }
  __syncthreads();
  for (int t = threadIdx.x; t < maxTiles; t += blockDim.x) {
    int r = t * TROWS, e = -1;
    for (int k = 0; k < KEXP; ++k)
      if (r >= soff[k] && r < soff[k + 1]) e = k;
    tileExpert[t] = e;
  }
}

__global__ void gather_kernel(const int* __restrict__ idxs,
                              const float* __restrict__ positions,
                              const float* __restrict__ viewdirs,
                              const float* __restrict__ features,
                              const int* __restrict__ offsets, int* __restrict__ cursors,
                              int* __restrict__ perm, _Float16* __restrict__ X, int n) {
  int i = blockIdx.x * 256 + threadIdx.x;
  if (i >= n) return;
  int e = idxs[i];
  int rank = atomicAdd(&cursors[e], 1);
  int row = offsets[e] + rank;
  perm[row] = i;

  __align__(16) _Float16 t[DINP];
  #pragma unroll
  for (int c = 0; c < DFEAT; ++c) t[c] = (_Float16)features[(size_t)i * DFEAT + c];
  float p[3] = { positions[i * 3], positions[i * 3 + 1], positions[i * 3 + 2] };
  float v[3] = { viewdirs[i * 3],  viewdirs[i * 3 + 1],  viewdirs[i * 3 + 2] };
  // posenc(pos, 2): identity(3) + sin(6) + cos(6) -> x[32..46]
  #pragma unroll
  for (int c = 0; c < 3; ++c) t[32 + c] = (_Float16)p[c];
  #pragma unroll
  for (int s = 0; s < 2; ++s) {
    float sc = (float)(1 << s);
    #pragma unroll
    for (int c = 0; c < 3; ++c) {
      t[35 + 3 * s + c] = (_Float16)__sinf(p[c] * sc);
      t[41 + 3 * s + c] = (_Float16)__cosf(p[c] * sc);
    }
  }
  // posenc(view, 4): identity(3) + sin(12) + cos(12) -> x[47..73]
  #pragma unroll
  for (int c = 0; c < 3; ++c) t[47 + c] = (_Float16)v[c];
  #pragma unroll
  for (int s = 0; s < 4; ++s) {
    float sc = (float)(1 << s);
    #pragma unroll
    for (int c = 0; c < 3; ++c) {
      t[50 + 3 * s + c] = (_Float16)__sinf(v[c] * sc);
      t[62 + 3 * s + c] = (_Float16)__cosf(v[c] * sc);
    }
  }
  #pragma unroll
  for (int c = DIN; c < DINP; ++c) t[c] = (_Float16)0.0f;

  uint4* dst = (uint4*)(X + (size_t)row * DINP);
  const uint4* src = (const uint4*)t;
  #pragma unroll
  for (int q = 0; q < DINP / 8; ++q) dst[q] = src[q];
}

__global__ void pad_kernel(const int* __restrict__ counts, const int* __restrict__ offsets,
                           int* __restrict__ perm, _Float16* __restrict__ X) {
  int j = blockIdx.x * 256 + threadIdx.x;
  if (j >= KEXP * TROWS) return;
  int e = j >> 7, i = j & 127;
  int row = offsets[e] + counts[e] + i;
  if (row < offsets[e + 1]) {
    perm[row] = -1;
    uint4 z; z.x = z.y = z.z = z.w = 0u;
    uint4* dst = (uint4*)(X + (size_t)row * DINP);
    #pragma unroll
    for (int q = 0; q < DINP / 8; ++q) dst[q] = z;
  }
}

// Transpose + f16-convert weights to [expert][dout][din_pad]
__global__ void weights_kernel(const float* __restrict__ W0, const float* __restrict__ W1,
                               const float* __restrict__ W2,
                               _Float16* __restrict__ W0t, _Float16* __restrict__ W1t,
                               _Float16* __restrict__ W2t) {
  int t = blockIdx.x * 256 + threadIdx.x;
  const int n0 = KEXP * WIDTH * DINP;    // 49152
  const int n1 = KEXP * WIDTH * WIDTH;   // 32768
  const int n2 = KEXP * DFEAT * WIDTH;   // 16384
  if (t < n0) {
    int e = t / (WIDTH * DINP), rem = t % (WIDTH * DINP);
    int r = rem / DINP, c = rem % DINP;
    W0t[t] = (_Float16)(c < DIN ? W0[((size_t)e * DIN + c) * WIDTH + r] : 0.0f);
  } else if (t < n0 + n1) {
    int u = t - n0;
    int e = u / (WIDTH * WIDTH), rem = u % (WIDTH * WIDTH);
    int r = rem / WIDTH, c = rem % WIDTH;
    W1t[u] = (_Float16)W1[((size_t)e * WIDTH + c) * WIDTH + r];
  } else if (t < n0 + n1 + n2) {
    int u = t - n0 - n1;
    int e = u / (DFEAT * WIDTH), rem = u % (DFEAT * WIDTH);
    int r = rem / WIDTH, c = rem % WIDTH;
    W2t[u] = (_Float16)W2[((size_t)e * WIDTH + c) * DFEAT + r];
  }
}

// ---------------- fused MoE MLP (WMMA) ----------------

__global__ __launch_bounds__(256) void moe_mlp_kernel(
    const _Float16* __restrict__ X,
    const _Float16* __restrict__ W0t, const _Float16* __restrict__ W1t,
    const _Float16* __restrict__ W2t,
    const float* __restrict__ b0, const float* __restrict__ b1, const float* __restrict__ b2,
    const int* __restrict__ tileExpert, const int* __restrict__ perm,
    float* __restrict__ out) {
  __shared__ __align__(16) _Float16 sW0[WIDTH * DINP];    // 12 KB
  __shared__ __align__(16) _Float16 sW1[WIDTH * WIDTH];   //  8 KB
  __shared__ __align__(16) _Float16 sW2[DFEAT * WIDTH];   //  4 KB
  __shared__ __align__(16) _Float16 sH0[TROWS * WIDTH];   // 16 KB
  __shared__ __align__(16) _Float16 sH1[TROWS * WIDTH];   // 16 KB
  __shared__ float sB0[WIDTH];
  __shared__ float sB1[WIDTH];
  __shared__ float sB2[DFEAT];

  const int e = tileExpert[blockIdx.x];
  if (e < 0) return;                       // uniform across the block
  const int tid = threadIdx.x;
  const int w = tid >> 5, l = tid & 31;
  const int rlo = l & 15, rhi = l >> 4;
  const int rowBase = w * 16;                       // each wave owns 16 rows
  const size_t tileRow0 = (size_t)blockIdx.x * TROWS;

  // Prefetch this wave's A rows (192 B each) into cache while weights stage.
  const _Float16* arow0 = X + (tileRow0 + rowBase + rlo) * DINP;
  __builtin_prefetch((const void*)arow0, 0, 3);
  __builtin_prefetch((const void*)((const char*)arow0 + 128), 0, 3);

  // ---- stage this expert's weights/biases into LDS ----
#if ATHENA_ASYNC_LDS
  {
    const char* gW0 = (const char*)(W0t + (size_t)e * (WIDTH * DINP));
    const char* gW1 = (const char*)(W1t + (size_t)e * (WIDTH * WIDTH));
    const char* gW2 = (const char*)(W2t + (size_t)e * (DFEAT * WIDTH));
    for (int i = tid; i < WIDTH * DINP / 8; i += 256)
      async_copy16(gW0 + (size_t)i * 16, (char*)sW0 + (size_t)i * 16);
    for (int i = tid; i < WIDTH * WIDTH / 8; i += 256)
      async_copy16(gW1 + (size_t)i * 16, (char*)sW1 + (size_t)i * 16);
    for (int i = tid; i < DFEAT * WIDTH / 8; i += 256)
      async_copy16(gW2 + (size_t)i * 16, (char*)sW2 + (size_t)i * 16);
  }
#else
  {
    const uint4* g = (const uint4*)(W0t + (size_t)e * (WIDTH * DINP));
    uint4* s = (uint4*)sW0;
    for (int i = tid; i < WIDTH * DINP / 8; i += 256) s[i] = g[i];
    g = (const uint4*)(W1t + (size_t)e * (WIDTH * WIDTH)); s = (uint4*)sW1;
    for (int i = tid; i < WIDTH * WIDTH / 8; i += 256) s[i] = g[i];
    g = (const uint4*)(W2t + (size_t)e * (DFEAT * WIDTH)); s = (uint4*)sW2;
    for (int i = tid; i < DFEAT * WIDTH / 8; i += 256) s[i] = g[i];
  }
#endif
  if (tid < WIDTH) { sB0[tid] = b0[e * WIDTH + tid]; sB1[tid] = b1[e * WIDTH + tid]; }
  if (tid < DFEAT) sB2[tid] = b2[e * DFEAT + tid];
#if ATHENA_ASYNC_LDS
  wait_async0();   // our async deposits done; barrier makes all waves' visible
#endif
  __syncthreads();

  // ---- layer 0: [128 x 96] x [96 x 64] ----
  v8f acc0[4] = {};
  {
    const _Float16* arow = arow0;
    #pragma unroll
    for (int ks = 0; ks < DINP / 32; ++ks) {
      v16h a = load_frag_A(arow, ks * 32, rhi);
      #pragma unroll
      for (int nt = 0; nt < 4; ++nt) {
        v16h b = load_frag_B(sW0, DINP, nt * 16 + rlo, ks * 32, rhi);
        acc0[nt] = __builtin_amdgcn_wmma_f32_16x16x32_f16(
            false, a, false, b, (short)0, acc0[nt], false, false);
      }
    }
    #pragma unroll
    for (int nt = 0; nt < 4; ++nt) {
      int col = nt * 16 + rlo;
      float bias = sB0[col];
      #pragma unroll
      for (int i = 0; i < 8; ++i) {
        float vv = acc0[nt][i] + bias;
        vv = vv > 0.0f ? vv : 0.0f;                         // ReLU
        sH0[(rowBase + rhi * 8 + i) * WIDTH + col] = (_Float16)vv;
      }
    }
  }
  __syncthreads();

  // ---- layer 1: [128 x 64] x [64 x 64] ----
  v8f acc1[4] = {};
  {
    const _Float16* arow = sH0 + (rowBase + rlo) * WIDTH;
    #pragma unroll
    for (int ks = 0; ks < WIDTH / 32; ++ks) {
      v16h a = load_frag_A(arow, ks * 32, rhi);
      #pragma unroll
      for (int nt = 0; nt < 4; ++nt) {
        v16h b = load_frag_B(sW1, WIDTH, nt * 16 + rlo, ks * 32, rhi);
        acc1[nt] = __builtin_amdgcn_wmma_f32_16x16x32_f16(
            false, a, false, b, (short)0, acc1[nt], false, false);
      }
    }
    #pragma unroll
    for (int nt = 0; nt < 4; ++nt) {
      int col = nt * 16 + rlo;
      float bias = sB1[col];
      #pragma unroll
      for (int i = 0; i < 8; ++i) {
        float vv = acc1[nt][i] + bias;
        vv = vv > 0.0f ? vv : 0.0f;
        sH1[(rowBase + rhi * 8 + i) * WIDTH + col] = (_Float16)vv;
      }
    }
  }
  __syncthreads();

  // ---- layer 2: [128 x 64] x [64 x 32], scatter via perm ----
  v8f acc2[2] = {};
  {
    const _Float16* arow = sH1 + (rowBase + rlo) * WIDTH;
    #pragma unroll
    for (int ks = 0; ks < WIDTH / 32; ++ks) {
      v16h a = load_frag_A(arow, ks * 32, rhi);
      #pragma unroll
      for (int nt = 0; nt < 2; ++nt) {
        v16h b = load_frag_B(sW2, WIDTH, nt * 16 + rlo, ks * 32, rhi);
        acc2[nt] = __builtin_amdgcn_wmma_f32_16x16x32_f16(
            false, a, false, b, (short)0, acc2[nt], false, false);
      }
    }
    #pragma unroll
    for (int nt = 0; nt < 2; ++nt) {
      int col = nt * 16 + rlo;
      float bias = sB2[col];
      #pragma unroll
      for (int i = 0; i < 8; ++i) {
        int g = perm[tileRow0 + rowBase + rhi * 8 + i];
        if (g >= 0) out[(size_t)g * DFEAT + col] = acc2[nt][i] + bias;
      }
    }
  }
}

// ---------------- launch ----------------

extern "C" void kernel_launch(void* const* d_in, const int* in_sizes, int n_in,
                              void* d_out, int out_size, void* d_ws, size_t ws_size,
                              hipStream_t stream) {
  const int*   idxs      = (const int*)d_in[0];
  const float* positions = (const float*)d_in[1];
  const float* viewdirs  = (const float*)d_in[2];
  const float* features  = (const float*)d_in[3];
  const float* W0        = (const float*)d_in[4];
  const float* b0        = (const float*)d_in[5];
  const float* W1        = (const float*)d_in[6];
  const float* b1        = (const float*)d_in[7];
  const float* W2        = (const float*)d_in[8];
  const float* b2        = (const float*)d_in[9];
  float* out = (float*)d_out;

  const int n = in_sizes[0];                              // idxs is [N,1]
  const int maxTiles  = (n + TROWS - 1) / TROWS + KEXP;   // padded upper bound
  const int paddedMax = maxTiles * TROWS;

  // workspace layout
  char* ws = (char*)d_ws;
  int* counts     = (int*)ws;                 // 8
  int* cursors    = counts + KEXP;            // 8
  int* offsets    = cursors + KEXP;           // 9
  int* tileExpert = offsets + KEXP + 1;       // maxTiles
  int* perm       = tileExpert + maxTiles;    // paddedMax
  size_t off = (size_t)(2 * KEXP + KEXP + 1 + maxTiles + paddedMax) * sizeof(int);
  off = (off + 255) & ~(size_t)255;
  _Float16* X   = (_Float16*)(ws + off); off += (size_t)paddedMax * DINP * sizeof(_Float16);
  off = (off + 255) & ~(size_t)255;
  _Float16* W0t = (_Float16*)(ws + off); off += (size_t)KEXP * WIDTH * DINP * sizeof(_Float16);
  _Float16* W1t = (_Float16*)(ws + off); off += (size_t)KEXP * WIDTH * WIDTH * sizeof(_Float16);
  _Float16* W2t = (_Float16*)(ws + off); off += (size_t)KEXP * DFEAT * WIDTH * sizeof(_Float16);

  const int nBlocks = (n + 255) / 256;
  zero_kernel<<<1, 32, 0, stream>>>(counts);
  count_kernel<<<nBlocks, 256, 0, stream>>>(idxs, counts, n);
  offsets_kernel<<<1, 256, 0, stream>>>(counts, offsets, tileExpert, maxTiles);
  pad_kernel<<<(KEXP * TROWS + 255) / 256, 256, 0, stream>>>(counts, offsets, perm, X);
  gather_kernel<<<nBlocks, 256, 0, stream>>>(idxs, positions, viewdirs, features,
                                             offsets, cursors, perm, X, n);
  const int wTotal = KEXP * (WIDTH * DINP + WIDTH * WIDTH + DFEAT * WIDTH);
  weights_kernel<<<(wTotal + 255) / 256, 256, 0, stream>>>(W0, W1, W2, W0t, W1t, W2t);
  moe_mlp_kernel<<<maxTiles, 256, 0, stream>>>(X, W0t, W1t, W2t, b0, b1, b2,
                                               tileExpert, perm, out);
}
